// CT_Attention_27831388078459
// MI455X (gfx1250) — matrix operand order repaired and checked by reading, not compiled, not run on hardware
//
#include <hip/hip_runtime.h>
#include <hip/hip_bf16.h>

typedef __attribute__((ext_vector_type(16))) _Float16 v16h;
typedef __attribute__((ext_vector_type(8)))  float    v8f;

#define DEV static __device__ __forceinline__

constexpr int DIMC  = 256;
constexpr int HEADS = 8;
constexpr int DHEAD = 64;
constexpr int INNER = 512;   // HEADS*DHEAD
constexpr int HW    = 1024;  // 32*32 (q tokens)
constexpr int HWKV  = 256;   // 16*16 (kv tokens)
constexpr float SCL    = 0.125f;  // 64^-0.5
constexpr float BN_EPS = 1e-5f;
constexpr int LPITCH = 264;  // LDS row pitch (halfs), padded

// ---------- WMMA fragment helpers (contiguous k = 16*(lane/16) + s order, wave32) ----------

// A fragment 16x32 f16 from row-major [16, >=32]; base points at element [0, k0]
DEV v16h load_a16(const _Float16* base, int ld) {
  int l = threadIdx.x & 31;
  const _Float16* p = base + (l & 15) * ld + ((l >> 4) << 4);
  v16h r;
#pragma unroll
  for (int e = 0; e < 8; ++e) { r[e] = p[e]; r[e + 8] = p[e + 8]; }
  return r;
}

// B fragment 32x16 f16 from [N,K]-row-major storage (K contiguous); base at [n0, k0]
DEV v16h load_b16(const _Float16* base, int ld) {
  int l = threadIdx.x & 31;
  const _Float16* p = base + (l & 15) * ld + ((l >> 4) << 4);
  v16h r;
#pragma unroll
  for (int e = 0; e < 16; ++e) r[e] = p[e];
  return r;
}

DEV v8f wmma(v16h a, v16h b, v8f c) {
  return __builtin_amdgcn_wmma_f32_16x16x32_f16(false, a, false, b, (short)0, c,
                                                false, false);
}

DEV v8f vzero() { v8f z = {0.f,0.f,0.f,0.f,0.f,0.f,0.f,0.f}; return z; }

// 64x32 macro-tile accumulate: A row-major [64, K] at a0, B storage [32, K] at b0.
// 4x2 register blocking -> 8 wmma per K-step from 4+2 fragment loads.
// 8 accumulators (64 VGPRs) + 6 fragments (48 VGPRs): fits without spills.
template <int K>
DEV void gemm64x32(const _Float16* __restrict__ a0, const _Float16* __restrict__ b0,
                   v8f (&acc)[4][2]) {
#pragma unroll
  for (int mi = 0; mi < 4; ++mi)
#pragma unroll
    for (int ni = 0; ni < 2; ++ni) acc[mi][ni] = vzero();

#pragma unroll
  for (int kt = 0; kt < K; kt += 32) {
    __builtin_prefetch(b0 + kt + 256, 0, 1);  // stream hint on B operand
    v16h af[4], bf[2];
#pragma unroll
    for (int mi = 0; mi < 4; ++mi) af[mi] = load_a16(a0 + (size_t)(mi * 16) * K + kt, K);
#pragma unroll
    for (int ni = 0; ni < 2; ++ni) bf[ni] = load_b16(b0 + (size_t)(ni * 16) * K + kt, K);
#pragma unroll
    for (int mi = 0; mi < 4; ++mi)
#pragma unroll
      for (int ni = 0; ni < 2; ++ni) acc[mi][ni] = wmma(af[mi], bf[ni], acc[mi][ni]);
  }
}

// ---------- K0: weight f32 -> f16 ----------
__global__ void k_cvt_weights(const float* __restrict__ wq, const float* __restrict__ wkv,
                              const float* __restrict__ wout,
                              _Float16* __restrict__ fq, _Float16* __restrict__ fkv,
                              _Float16* __restrict__ fout) {
  int i = blockIdx.x * blockDim.x + threadIdx.x;
  if (i < INNER * DIMC)     fq[i]   = (_Float16)wq[i];
  if (i < 2 * INNER * DIMC) fkv[i]  = (_Float16)wkv[i];
  if (i < DIMC * INNER)     fout[i] = (_Float16)wout[i];
}

// ---------- K1: depthwise 3x3 + BN(eval) -> f16 activations [s,b,hw,c] ----------
__global__ void k_dw_bn(const float* __restrict__ x, const float* __restrict__ y,
                        const float* __restrict__ w, const float* __restrict__ gm,
                        const float* __restrict__ bt, const float* __restrict__ mn,
                        const float* __restrict__ vr, _Float16* __restrict__ out,
                        int stride, int OS) {
  int idx = blockIdx.x * blockDim.x + threadIdx.x;
  int c   = idx & (DIMC - 1);
  int t   = idx >> 8;
  int ohw = t % (OS * OS);
  int sb  = t / (OS * OS);
  int b = sb & 7, s = sb >> 3;
  int oy = ohw / OS, ox = ohw % OS;
  const float* in = (s ? y : x) + ((size_t)(b * DIMC + c)) * 32 * 32;
  const float* wc = w + c * 9;
  float acc = 0.f;
#pragma unroll
  for (int ky = 0; ky < 3; ++ky) {
    int iy = oy * stride + ky - 1;
    if ((unsigned)iy < 32u) {
#pragma unroll
      for (int kx = 0; kx < 3; ++kx) {
        int ix = ox * stride + kx - 1;
        if ((unsigned)ix < 32u) acc += in[iy * 32 + ix] * wc[ky * 3 + kx];
      }
    }
  }
  float sc = gm[c] * rsqrtf(vr[c] + BN_EPS);
  out[idx] = (_Float16)((acc - mn[c]) * sc + bt[c]);
}

// ---------- K2: Q = Wq x Hq, stored transposed QT[z][i(1024)][o(512)] f16 ----------
__global__ __launch_bounds__(32, 1) void k_gemm_q(const _Float16* __restrict__ W,
                                                  const _Float16* __restrict__ H,
                                                  _Float16* __restrict__ QT) {
  int mt = blockIdx.x, nt = blockIdx.y, z = blockIdx.z;  // (8, 32, 16)
  v8f acc[4][2];
  gemm64x32<DIMC>(W + (size_t)(mt * 64) * DIMC,
                  H + (size_t)(z * HW + nt * 32) * DIMC, acc);
  int l = threadIdx.x & 31;
  int nl = l & 15, mh = (l >> 4) << 3;
#pragma unroll
  for (int ni = 0; ni < 2; ++ni) {
    int n = nt * 32 + ni * 16 + nl;
    _Float16* o = QT + (size_t)(z * HW + n) * INNER + mt * 64 + mh;
#pragma unroll
    for (int mi = 0; mi < 4; ++mi)
#pragma unroll
      for (int r = 0; r < 8; ++r) o[mi * 16 + r] = (_Float16)acc[mi][ni][r];
  }
}

// ---------- K3: KV = Wkv x Hkv; K transposed KT[z][j][o], V direct V[z][o][j] ----------
__global__ __launch_bounds__(32, 1) void k_gemm_kv(const _Float16* __restrict__ W,
                                                   const _Float16* __restrict__ H,
                                                   _Float16* __restrict__ KT,
                                                   _Float16* __restrict__ Vb) {
  int mt = blockIdx.x, nt = blockIdx.y, z = blockIdx.z;  // (16, 8, 16)
  v8f acc[4][2];
  gemm64x32<DIMC>(W + (size_t)(mt * 64) * DIMC,
                  H + (size_t)(z * HWKV + nt * 32) * DIMC, acc);
  int l = threadIdx.x & 31;
  int nl = l & 15, mh = (l >> 4) << 3;
  int m0 = mt * 64;
  if (m0 < INNER) {  // K half: transposed for B-fragment reuse in dots GEMM
#pragma unroll
    for (int ni = 0; ni < 2; ++ni) {
      int n = nt * 32 + ni * 16 + nl;
      _Float16* o = KT + (size_t)(z * HWKV + n) * INNER + m0 + mh;
#pragma unroll
      for (int mi = 0; mi < 4; ++mi)
#pragma unroll
        for (int r = 0; r < 8; ++r) o[mi * 16 + r] = (_Float16)acc[mi][ni][r];
    }
  } else {           // V half: direct [o, j] for B-fragment of attn@V^T
#pragma unroll
    for (int mi = 0; mi < 4; ++mi)
#pragma unroll
      for (int r = 0; r < 8; ++r) {
        int m = m0 - INNER + mi * 16 + mh + r;
        _Float16* o = Vb + (size_t)(z * INNER + m) * HWKV + nt * 32 + nl;
#pragma unroll
        for (int ni = 0; ni < 2; ++ni) o[ni * 16] = (_Float16)acc[mi][ni][r];
      }
  }
}

// ---------- K4: cross-attention per (i-block, head, z); 4 waves x 16 rows ----------
__global__ __launch_bounds__(128) void k_attn(const _Float16* __restrict__ QT,
                                              const _Float16* __restrict__ KT,
                                              const _Float16* __restrict__ Vb,
                                              _Float16* __restrict__ O) {
  __shared__ _Float16 lds[4 * 16 * LPITCH];
  int ib = blockIdx.x, h = blockIdx.y, z = blockIdx.z;
  int zz = z ^ 8;  // cross: swap x<->y stream, keep batch
  int w = threadIdx.x >> 5;
  int l = threadIdx.x & 31;
  int i0 = ib * 64 + w * 16;

  const _Float16* qbase = QT + (size_t)(z * HW + i0) * INNER + h * DHEAD;
  v16h aq0 = load_a16(qbase, INNER);
  v16h aq1 = load_a16(qbase + 32, INNER);

  v8f acc[16];
#pragma unroll
  for (int t = 0; t < 16; ++t) acc[t] = vzero();

  const _Float16* kbase = KT + (size_t)(zz * HWKV) * INNER + h * DHEAD;
#pragma unroll
  for (int t = 0; t < 16; ++t) {
    v16h b0 = load_b16(kbase + (size_t)(t * 16) * INNER, INNER);
    v16h b1 = load_b16(kbase + (size_t)(t * 16) * INNER + 32, INNER);
    acc[t] = wmma(aq0, b0, acc[t]);
    acc[t] = wmma(aq1, b1, acc[t]);
  }

  // softmax over j=256; row m = r + 8*(lane/16) lives across one 16-lane group
  _Float16* myl = lds + w * 16 * LPITCH;
#pragma unroll
  for (int r = 0; r < 8; ++r) {
    float mx = -3.0e38f;
#pragma unroll
    for (int t = 0; t < 16; ++t) mx = fmaxf(mx, acc[t][r]);
#pragma unroll
    for (int s = 8; s >= 1; s >>= 1) mx = fmaxf(mx, __shfl_xor(mx, s, 32));
    float sum = 0.f;
#pragma unroll
    for (int t = 0; t < 16; ++t) {
      float e = __expf((acc[t][r] - mx) * SCL);
      acc[t][r] = e;
      sum += e;
    }
#pragma unroll
    for (int s = 8; s >= 1; s >>= 1) sum += __shfl_xor(sum, s, 32);
    float inv = 1.f / sum;
    int row = r + ((l >> 4) << 3);
    int col = l & 15;
#pragma unroll
    for (int t = 0; t < 16; ++t)
      myl[row * LPITCH + t * 16 + col] = (_Float16)(acc[t][r] * inv);
  }
  __syncthreads();

  // out^T[i, d] = attn[i, j] x V^T[j, d];  B from Vb rows [d, j] (j contiguous)
  const _Float16* vb = Vb + (size_t)(zz * INNER + h * DHEAD) * HWKV;
  _Float16* ob = O + (size_t)(z * HW + i0) * INNER + h * DHEAD;
#pragma unroll
  for (int nt2 = 0; nt2 < 4; ++nt2) {
    v8f a2 = vzero();
#pragma unroll
    for (int kt = 0; kt < HWKV; kt += 32) {
      v16h af = load_a16(myl + kt, LPITCH);
      v16h bf = load_b16(vb + (size_t)(nt2 * 16) * HWKV + kt, HWKV);
      a2 = wmma(af, bf, a2);
    }
    int n = nt2 * 16 + (l & 15);
    int mb = (l >> 4) << 3;
#pragma unroll
    for (int r = 0; r < 8; ++r)
      ob[(size_t)(mb + r) * INNER + n] = (_Float16)a2[r];
  }
}

// ---------- K5: final projection + bias, f32 out [s,b,256,1024] ----------
__global__ __launch_bounds__(32, 1) void k_gemm_out(const _Float16* __restrict__ W,
                                                    const _Float16* __restrict__ H,
                                                    const float* __restrict__ bias,
                                                    float* __restrict__ out) {
  int mt = blockIdx.x, nt = blockIdx.y, z = blockIdx.z;  // (4, 32, 16)
  v8f acc[4][2];
  gemm64x32<INNER>(W + (size_t)(mt * 64) * INNER,
                   H + (size_t)(z * HW + nt * 32) * INNER, acc);
  int l = threadIdx.x & 31;
  int nl = l & 15, mh = (l >> 4) << 3;
  float* o = out + (size_t)z * DIMC * HW;
#pragma unroll
  for (int mi = 0; mi < 4; ++mi)
#pragma unroll
    for (int r = 0; r < 8; ++r) {
      int m = mt * 64 + mi * 16 + mh + r;
      float bs = bias[m];
#pragma unroll
      for (int ni = 0; ni < 2; ++ni)
        o[(size_t)m * HW + nt * 32 + ni * 16 + nl] = acc[mi][ni][r] + bs;
    }
}

extern "C" void kernel_launch(void* const* d_in, const int* in_sizes, int n_in,
                              void* d_out, int out_size, void* d_ws, size_t ws_size,
                              hipStream_t stream) {
  const float* x       = (const float*)d_in[0];
  const float* y       = (const float*)d_in[1];
  const float* q_dw_w  = (const float*)d_in[2];
  const float* q_g     = (const float*)d_in[3];
  const float* q_b     = (const float*)d_in[4];
  const float* q_m     = (const float*)d_in[5];
  const float* q_v     = (const float*)d_in[6];
  const float* q_pw_w  = (const float*)d_in[7];
  const float* kv_dw_w = (const float*)d_in[8];
  const float* kv_g    = (const float*)d_in[9];
  const float* kv_b    = (const float*)d_in[10];
  const float* kv_m    = (const float*)d_in[11];
  const float* kv_v    = (const float*)d_in[12];
  const float* kv_pw_w = (const float*)d_in[13];
  const float* out_w   = (const float*)d_in[14];
  const float* out_b   = (const float*)d_in[15];

  char* ws = (char*)d_ws;
  size_t off = 0;
  auto carve = [&](size_t halfs) {
    _Float16* p = (_Float16*)(ws + off);
    off += halfs * sizeof(_Float16);
    return p;
  };
  _Float16* Wq  = carve((size_t)INNER * DIMC);
  _Float16* Wkv = carve((size_t)2 * INNER * DIMC);
  _Float16* Wo  = carve((size_t)DIMC * INNER);
  _Float16* hq  = carve((size_t)16 * HW * DIMC);
  _Float16* hkv = carve((size_t)16 * HWKV * DIMC);
  _Float16* QT  = carve((size_t)16 * HW * INNER);
  _Float16* KT  = carve((size_t)16 * HWKV * INNER);
  _Float16* Vb  = carve((size_t)16 * INNER * HWKV);
  _Float16* AO  = carve((size_t)16 * HW * INNER);

  k_cvt_weights<<<(2 * INNER * DIMC + 255) / 256, 256, 0, stream>>>(
      q_pw_w, kv_pw_w, out_w, Wq, Wkv, Wo);

  k_dw_bn<<<(16 * HW * DIMC) / 256, 256, 0, stream>>>(
      x, y, q_dw_w, q_g, q_b, q_m, q_v, hq, 1, 32);
  k_dw_bn<<<(16 * HWKV * DIMC) / 256, 256, 0, stream>>>(
      x, y, kv_dw_w, kv_g, kv_b, kv_m, kv_v, hkv, 2, 16);

  k_gemm_q<<<dim3(INNER / 64, HW / 32, 16), 32, 0, stream>>>(Wq, hq, QT);
  k_gemm_kv<<<dim3(2 * INNER / 64, HWKV / 32, 16), 32, 0, stream>>>(Wkv, hkv, KT, Vb);

  k_attn<<<dim3(HW / 64, HEADS, 16), 128, 0, stream>>>(QT, KT, Vb, AO);

  k_gemm_out<<<dim3(DIMC / 64, HW / 32, 16), 32, 0, stream>>>(Wo, AO, out_b,
                                                              (float*)d_out);
}
